// ReverseVideoDecoder_42606075576981
// MI455X (gfx1250) — matrix operand order repaired and checked by reading, not compiled, not run on hardware
//
#include <hip/hip_runtime.h>
#include <hip/hip_bf16.h>
#include <math.h>

// ---------------- problem constants (match reference) ----------------
#define RH 1024
#define RC 1024
#define RV 2048
#define RS 60
#define RT 40
#define RB 128
#define KSPLIT 256            // K-elements per split-K slice (8 wmma k-steps)
#define ALDA   264            // padded LDS row stride (bf16 elems): 528B = 132 dw, 132%64=4

typedef __attribute__((ext_vector_type(16))) __bf16 v16bf;
typedef __attribute__((ext_vector_type(8)))  __bf16 v8bf;
typedef __attribute__((ext_vector_type(8)))  float  v8f;

// =====================================================================
// Split-K GEMM, bf16 A and bf16 B(N,K):  Cpart[z] = A * B^T over K-slice z.
// Block = 128 threads (4 waves); each wave owns a 16(M) x 64(N) tile.
// grid = (N/256, M/16, K/KSPLIT).  M%16==0, N%256==0, K%KSPLIT==0.
// The 16x256 A slice (shared by all 4 waves) is staged into LDS ONCE via
// GLOBAL_LOAD_ASYNC_TO_LDS_B128 (ASYNCcnt + s_wait_asynccnt + barrier);
// the k-loop is then ds_load (A) + global_load_b128 (B) + WMMA.
// =====================================================================
__global__ __launch_bounds__(128)
void gemm_bf16a_splitk_kernel(const __bf16* __restrict__ A,
                              const __bf16* __restrict__ Bw,
                              float* __restrict__ Cpart,
                              int M, int N, int K)
{
    __shared__ __bf16 tile[16 * ALDA];     // 8448 B

    const int tid  = threadIdx.x;
    const int lane = tid & 31;
    const int wave = tid >> 5;
    const int l15  = lane & 15;
    const int hk   = lane >> 4;
    const int n0   = blockIdx.x * 256 + wave * 64;
    const int m0   = blockIdx.y * 16;
    const int kb   = blockIdx.z * KSPLIT;

    // ---- async-stage A slice: 16 rows x 256 bf16 = 512 x 16B chunks ----
    // thread t copies chunks [4t, 4t+3]; 4 chunks never straddle a row.
    {
        const int chunk0 = tid * 4;
        const int row    = chunk0 >> 5;          // 32 chunks per row
        const int cc     = chunk0 & 31;
        const __bf16* gbase = A + (size_t)(m0 + row) * K + kb + cc * 8;
        unsigned lbase = (unsigned)(uintptr_t)&tile[row * ALDA + cc * 8];
#pragma unroll
        for (int c = 0; c < 4; ++c) {
            asm volatile("global_load_async_to_lds_b128 %0, %1, off"
                         :: "v"(lbase + c * 16), "v"(gbase + c * 8)
                         : "memory");
        }
        asm volatile("s_wait_asynccnt 0x0" ::: "memory");
    }
    __syncthreads();

    v8f acc[4] = {};

    for (int k0 = 0; k0 < KSPLIT; k0 += 32) {
        // 16-bit A 16x32 fragment from LDS (ds_load_b128 x2 per lane)
        const __bf16* al = &tile[l15 * ALDA + k0 + hk * 8];
        v8bf lo = *(const v8bf*)(al);
        v8bf hi = *(const v8bf*)(al + 16);
        v16bf a = __builtin_shufflevector(lo, hi,
                      0, 1, 2, 3, 4, 5, 6, 7, 8, 9, 10, 11, 12, 13, 14, 15);
#pragma unroll
        for (int j = 0; j < 4; ++j) {
            const __bf16* brow = Bw + (size_t)(n0 + j * 16 + l15) * K + kb + k0 + hk * 8;
            v8bf blo = *(const v8bf*)(brow);
            v8bf bhi = *(const v8bf*)(brow + 16);
            v16bf b = __builtin_shufflevector(blo, bhi,
                      0, 1, 2, 3, 4, 5, 6, 7, 8, 9, 10, 11, 12, 13, 14, 15);
            __builtin_prefetch(brow + 32, 0, 1);
#if defined(__HIP_DEVICE_COMPILE__)
            acc[j] = __builtin_amdgcn_wmma_f32_16x16x32_bf16(
                false, a, false, b, (short)0, acc[j], false, false);
#endif
        }
    }

    float* out = Cpart + (size_t)blockIdx.z * M * N;
#pragma unroll
    for (int j = 0; j < 4; ++j) {
        const int col = n0 + j * 16 + l15;
#pragma unroll
        for (int r = 0; r < 8; ++r) {
            const int row = m0 + hk * 8 + r;
            out[(size_t)row * N + col] = acc[j][r];
        }
    }
}

// =====================================================================
// One-shot GEMM with f32 A (used only for ctx_proj, M=7680): direct f32 out.
// =====================================================================
__global__ __launch_bounds__(128)
void gemm_f32a_kernel(const float* __restrict__ A,
                      const __bf16* __restrict__ Bw,
                      float* __restrict__ Cout,
                      int M, int N, int K)
{
    const int lane = threadIdx.x & 31;
    const int wave = threadIdx.x >> 5;
    const int l15  = lane & 15;
    const int hk   = lane >> 4;
    const int n0   = blockIdx.x * 256 + wave * 64;
    const int m0   = blockIdx.y * 16;
    const int m    = m0 + l15;

    v8f acc[4] = {};
    const float* arow = A + (size_t)m * K + hk * 8;

    for (int k0 = 0; k0 < K; k0 += 32) {
        float4 f0 = *(const float4*)(arow + k0);
        float4 f1 = *(const float4*)(arow + k0 + 4);
        float4 f2 = *(const float4*)(arow + k0 + 16);
        float4 f3 = *(const float4*)(arow + k0 + 20);
        v16bf a;
        a[0]  = (__bf16)f0.x; a[1]  = (__bf16)f0.y; a[2]  = (__bf16)f0.z; a[3]  = (__bf16)f0.w;
        a[4]  = (__bf16)f1.x; a[5]  = (__bf16)f1.y; a[6]  = (__bf16)f1.z; a[7]  = (__bf16)f1.w;
        a[8]  = (__bf16)f2.x; a[9]  = (__bf16)f2.y; a[10] = (__bf16)f2.z; a[11] = (__bf16)f2.w;
        a[12] = (__bf16)f3.x; a[13] = (__bf16)f3.y; a[14] = (__bf16)f3.z; a[15] = (__bf16)f3.w;
#pragma unroll
        for (int j = 0; j < 4; ++j) {
            const __bf16* brow = Bw + (size_t)(n0 + j * 16 + l15) * K + k0 + hk * 8;
            v8bf blo = *(const v8bf*)(brow);
            v8bf bhi = *(const v8bf*)(brow + 16);
            v16bf b = __builtin_shufflevector(blo, bhi,
                      0, 1, 2, 3, 4, 5, 6, 7, 8, 9, 10, 11, 12, 13, 14, 15);
            __builtin_prefetch(brow + 32, 0, 1);
#if defined(__HIP_DEVICE_COMPILE__)
            acc[j] = __builtin_amdgcn_wmma_f32_16x16x32_bf16(
                false, a, false, b, (short)0, acc[j], false, false);
#endif
        }
    }
#pragma unroll
    for (int j = 0; j < 4; ++j) {
        const int col = n0 + j * 16 + l15;
#pragma unroll
        for (int r = 0; r < 8; ++r) {
            const int row = m0 + hk * 8 + r;
            Cout[(size_t)row * N + col] = acc[j][r];
        }
    }
}

// =====================================================================
// Split-K combine: out = act( sum_z Cpart[z] + bias ); optional f32/bf16 out.
// =====================================================================
__global__ void combine_kernel(const float* __restrict__ Cpart, int SK, int MN,
                               const float* __restrict__ bias, int N, int act,
                               float* __restrict__ outF, __bf16* __restrict__ outB)
{
    int idx = blockIdx.x * blockDim.x + threadIdx.x;
    if (idx >= MN) return;
    float s = 0.0f;
    for (int z = 0; z < SK; ++z) s += Cpart[(size_t)z * MN + idx];
    if (bias) s += bias[idx % N];
    if (act == 1) s = tanhf(s);
    if (outF) outF[idx] = s;
    if (outB) outB[idx] = (__bf16)s;
}

// ---------------- weight prep ----------------
__global__ void conv_f32_to_bf16_kernel(const float* __restrict__ src,
                                        __bf16* __restrict__ dst, int n)
{
    int i = blockIdx.x * blockDim.x + threadIdx.x;
    if (i < n) dst[i] = (__bf16)src[i];
}

// src is (K,N) row-major f32 -> dst is (N,K) row-major bf16
__global__ void transpose_to_bf16_kernel(const float* __restrict__ src,
                                         __bf16* __restrict__ dst, int K, int N)
{
    int i = blockIdx.x * blockDim.x + threadIdx.x;
    if (i >= N * K) return;
    int n = i / K;
    int k = i - n * K;
    dst[i] = (__bf16)src[(size_t)k * N + n];
}

__global__ void zero_f32_kernel(float* __restrict__ p, int n)
{
    int i = blockIdx.x * blockDim.x + threadIdx.x;
    if (i < n) p[i] = 0.0f;
}

__global__ void zero_bf16_kernel(__bf16* __restrict__ p, int n)
{
    int i = blockIdx.x * blockDim.x + threadIdx.x;
    if (i < n) p[i] = (__bf16)0.0f;
}

// ---------------- GRU pointwise (emits f32 + bf16 new hidden state) --------
__global__ void gru_pointwise_kernel(const float* __restrict__ gi,
                                     const float* __restrict__ gh,
                                     const float* __restrict__ hin,
                                     float* __restrict__ hout,
                                     __bf16* __restrict__ hout_bf)
{
    int idx = blockIdx.x * blockDim.x + threadIdx.x;   // over B*H
    int b = idx >> 10;           // H = 1024
    int j = idx & 1023;
    size_t base = (size_t)b * (3 * RH);
    float ir = gi[base + j], iz = gi[base + RH + j], in_ = gi[base + 2 * RH + j];
    float hr = gh[base + j], hz = gh[base + RH + j], hn  = gh[base + 2 * RH + j];
    float r = 1.0f / (1.0f + expf(-(ir + hr)));
    float z = 1.0f / (1.0f + expf(-(iz + hz)));
    float n = tanhf(in_ + r * hn);
    float hv = (1.0f - z) * n + z * hin[idx];
    hout[idx] = hv;
    hout_bf[idx] = (__bf16)hv;
}

// ---------------- attention ----------------
__global__ __launch_bounds__(256)
void scores_kernel(const float* __restrict__ ctxp,
                   const float* __restrict__ hWh,
                   const float* __restrict__ w_mlp,
                   float* __restrict__ scores)
{
    __shared__ float sm[256];
    int sb = blockIdx.x;          // s*B + b
    int b  = sb & (RB - 1);
    const float* cp = ctxp + (size_t)sb * RC;
    const float* hw = hWh + (size_t)b * RC;
    float s = 0.0f;
    for (int c = threadIdx.x; c < RC; c += 256)
        s += tanhf(cp[c] + hw[c]) * w_mlp[c];
    sm[threadIdx.x] = s;
    __syncthreads();
    for (int off = 128; off > 0; off >>= 1) {
        if (threadIdx.x < off) sm[threadIdx.x] += sm[threadIdx.x + off];
        __syncthreads();
    }
    if (threadIdx.x == 0) scores[sb] = sm[0];
}

__global__ __launch_bounds__(64)
void softmax_kernel(const float* __restrict__ scores, float* __restrict__ alpha)
{
    __shared__ float sm[64];
    int b = blockIdx.x;
    int t = threadIdx.x;
    float v = (t < RS) ? scores[t * RB + b] : -1e30f;
    sm[t] = v;
    __syncthreads();
    for (int off = 32; off > 0; off >>= 1) {
        if (t < off) sm[t] = fmaxf(sm[t], sm[t + off]);
        __syncthreads();
    }
    float mx = sm[0];
    __syncthreads();
    float ev = (t < RS) ? expf(v - mx) : 0.0f;
    sm[t] = ev;
    __syncthreads();
    for (int off = 32; off > 0; off >>= 1) {
        if (t < off) sm[t] += sm[t + off];
        __syncthreads();
    }
    float denom = sm[0];
    if (t < RS) alpha[t * RB + b] = ev / denom;
}

// z_bf[b,c] = (bf16) sum_s alpha[s,b] * ctx[s,b,c]   (consumed only as GEMM A)
__global__ void zt_kernel(const float* __restrict__ alpha,
                          const float* __restrict__ ctx,
                          __bf16* __restrict__ z_bf)
{
    int idx = blockIdx.x * blockDim.x + threadIdx.x;   // over B*C
    int b = idx >> 10;
    int c = idx & 1023;
    float acc = 0.0f;
    for (int s = 0; s < RS; ++s)
        acc += alpha[s * RB + b] * ctx[((size_t)s * RB + b) * RC + c];
    z_bf[idx] = (__bf16)acc;
}

// ---------------- loss (deterministic two-stage reduction) ----------------
__global__ __launch_bounds__(256)
void loss_partial_kernel(const float* __restrict__ pv,
                         const float* __restrict__ yt,
                         float* __restrict__ partials)
{
    __shared__ float sm[256];
    int idx = blockIdx.x * 256 + threadIdx.x;          // B*V = 1024 blocks * 256
    float d = pv[idx] - yt[idx];
    sm[threadIdx.x] = d * d;
    __syncthreads();
    for (int off = 128; off > 0; off >>= 1) {
        if (threadIdx.x < off) sm[threadIdx.x] += sm[threadIdx.x + off];
        __syncthreads();
    }
    if (threadIdx.x == 0) partials[blockIdx.x] = sm[0];
}

__global__ __launch_bounds__(256)
void loss_accum_kernel(const float* __restrict__ partials, float* __restrict__ accum)
{
    __shared__ float sm[256];
    float s = 0.0f;
    for (int i = threadIdx.x; i < 1024; i += 256) s += partials[i];
    sm[threadIdx.x] = s;
    __syncthreads();
    for (int off = 128; off > 0; off >>= 1) {
        if (threadIdx.x < off) sm[threadIdx.x] += sm[threadIdx.x + off];
        __syncthreads();
    }
    if (threadIdx.x == 0) accum[0] += sm[0];
}

__global__ void write_out_kernel(const float* __restrict__ accum, float* __restrict__ out)
{
    if (threadIdx.x == 0 && blockIdx.x == 0) out[0] = accum[0];
}

// =====================================================================
extern "C" void kernel_launch(void* const* d_in, const int* in_sizes, int n_in,
                              void* d_out, int out_size, void* d_ws, size_t ws_size,
                              hipStream_t stream)
{
    (void)in_sizes; (void)n_in; (void)out_size; (void)ws_size;

    const float* ctx   = (const float*)d_in[0];
    const float* y     = (const float*)d_in[1];
    const float* W_emb = (const float*)d_in[2];
    const float* b_emb = (const float*)d_in[3];
    const float* Wih0  = (const float*)d_in[4];
    const float* Whh0  = (const float*)d_in[5];
    const float* bih0  = (const float*)d_in[6];
    const float* bhh0  = (const float*)d_in[7];
    const float* Wih1  = (const float*)d_in[8];
    const float* Whh1  = (const float*)d_in[9];
    const float* bih1  = (const float*)d_in[10];
    const float* bhh1  = (const float*)d_in[11];
    const float* Wc2c  = (const float*)d_in[12];
    const float* Wh2c  = (const float*)d_in[13];
    const float* w_mlp = (const float*)d_in[14];
    const float* Wout  = (const float*)d_in[15];
    const float* b_out = (const float*)d_in[16];

    // ---------------- workspace carve-up ----------------
    char* ws = (char*)d_ws;
    size_t off = 0;
    auto allocF = [&](size_t nelem) -> float* {
        float* p = (float*)(ws + off);
        off = (off + nelem * sizeof(float) + 255) & ~(size_t)255;
        return p;
    };
    auto allocB = [&](size_t nelem) -> __bf16* {
        __bf16* p = (__bf16*)(ws + off);
        off = (off + nelem * sizeof(__bf16) + 255) & ~(size_t)255;
        return p;
    };

    float* ctxp     = allocF((size_t)RS * RB * RC);     // ctx @ Wc2c (f32)
    float* pv       = allocF((size_t)RB * RV);          // f32 pv (for loss)
    float* h        = allocF((size_t)RB * RH);
    float* h1       = allocF((size_t)RB * RH);
    float* gi       = allocF((size_t)RB * 3 * RH);
    float* gh       = allocF((size_t)RB * 3 * RH);
    float* hWh      = allocF((size_t)RB * RC);
    float* scoresB  = allocF((size_t)RS * RB);
    float* alphaB   = allocF((size_t)RS * RB);
    float* partials = allocF(1024);
    float* accum    = allocF(1);
    float* Cpart    = allocF((size_t)8 * RB * 3 * RH);  // max SK*M*N = 8 slices safe

    __bf16* Wih0b  = allocB((size_t)3 * RH * RH);
    __bf16* Whh0b  = allocB((size_t)3 * RH * RH);
    __bf16* Wih1b  = allocB((size_t)3 * RH * RC);
    __bf16* Whh1b  = allocB((size_t)3 * RH * RH);
    __bf16* WembTb = allocB((size_t)RH * RV);   // (N=H, K=V)
    __bf16* WoutTb = allocB((size_t)RV * RH);   // (N=V, K=H)
    __bf16* Wc2cTb = allocB((size_t)RC * RC);   // (N=C, K=C)
    __bf16* Wh2cTb = allocB((size_t)RC * RH);   // (N=C, K=H)

    __bf16* pv_bf  = allocB((size_t)RB * RV);   // bf16 activation mirrors
    __bf16* e_bf   = allocB((size_t)RB * RH);
    __bf16* h_bf   = allocB((size_t)RB * RH);
    __bf16* h1_bf  = allocB((size_t)RB * RH);
    __bf16* z_bf   = allocB((size_t)RB * RC);

    // ---------------- helpers ----------------
    auto gemm_sk = [&](const __bf16* A, const __bf16* Bw, int M, int N, int K) {
        dim3 grid(N / 256, M / 16, K / KSPLIT);
        gemm_bf16a_splitk_kernel<<<grid, 128, 0, stream>>>(A, Bw, Cpart, M, N, K);
    };
    auto combine = [&](int SK, int M, int N, const float* bias, int act,
                       float* outF, __bf16* outB) {
        int MN = M * N;
        combine_kernel<<<(MN + 255) / 256, 256, 0, stream>>>(
            Cpart, SK, MN, bias, N, act, outF, outB);
    };
    auto conv = [&](const float* src, __bf16* dst, int n) {
        conv_f32_to_bf16_kernel<<<(n + 255) / 256, 256, 0, stream>>>(src, dst, n);
    };
    auto transp = [&](const float* src, __bf16* dst, int K, int N) {
        int n = N * K;
        transpose_to_bf16_kernel<<<(n + 255) / 256, 256, 0, stream>>>(src, dst, K, N);
    };

    // ---------------- prep (runs every call; deterministic) ----------------
    zero_f32_kernel<<<(RB * RH + 255) / 256, 256, 0, stream>>>(h, RB * RH);
    zero_bf16_kernel<<<(RB * RH + 255) / 256, 256, 0, stream>>>(h_bf, RB * RH);
    zero_bf16_kernel<<<(RB * RV + 255) / 256, 256, 0, stream>>>(pv_bf, RB * RV);
    zero_f32_kernel<<<1, 32, 0, stream>>>(accum, 1);

    conv(Wih0, Wih0b, 3 * RH * RH);
    conv(Whh0, Whh0b, 3 * RH * RH);
    conv(Wih1, Wih1b, 3 * RH * RC);
    conv(Whh1, Whh1b, 3 * RH * RH);
    transp(W_emb, WembTb, RV, RH);   // (V,H) -> (H,V)
    transp(Wout,  WoutTb, RH, RV);   // (H,V) -> (V,H)
    transp(Wc2c,  Wc2cTb, RC, RC);
    transp(Wh2c,  Wh2cTb, RH, RC);

    // ctx_proj = ctx(S*B, C) @ Wc2c  — plenty of parallelism, direct one-shot
    {
        dim3 grid(RC / 256, (RS * RB) / 16);
        gemm_f32a_kernel<<<grid, 128, 0, stream>>>(ctx, Wc2cTb, ctxp,
                                                   RS * RB, RC, RC);
    }

    // ---------------- the 40 sequential decode steps ----------------
    for (int t = 0; t < RT; ++t) {
        // e = tanh(pv @ W_emb + b_emb)          M=128 N=1024 K=2048, SK=8
        gemm_sk(pv_bf, WembTb, RB, RH, RV);
        combine(RV / KSPLIT, RB, RH, b_emb, 1, nullptr, e_bf);

        // GRU layer 0 gates                     M=128 N=3072 K=1024, SK=4
        gemm_sk(e_bf, Wih0b, RB, 3 * RH, RH);
        combine(RH / KSPLIT, RB, 3 * RH, bih0, 0, gi, nullptr);
        gemm_sk(h_bf, Whh0b, RB, 3 * RH, RH);
        combine(RH / KSPLIT, RB, 3 * RH, bhh0, 0, gh, nullptr);
        gru_pointwise_kernel<<<(RB * RH) / 256, 256, 0, stream>>>(gi, gh, h, h1, h1_bf);

        // attention
        gemm_sk(h1_bf, Wh2cTb, RB, RC, RH);
        combine(RH / KSPLIT, RB, RC, nullptr, 0, hWh, nullptr);
        scores_kernel<<<RS * RB, 256, 0, stream>>>(ctxp, hWh, w_mlp, scoresB);
        softmax_kernel<<<RB, 64, 0, stream>>>(scoresB, alphaB);
        zt_kernel<<<(RB * RC) / 256, 256, 0, stream>>>(alphaB, ctx, z_bf);

        // GRU layer 1 gates (h2 -> h)
        gemm_sk(z_bf, Wih1b, RB, 3 * RH, RC);
        combine(RC / KSPLIT, RB, 3 * RH, bih1, 0, gi, nullptr);
        gemm_sk(h1_bf, Whh1b, RB, 3 * RH, RH);
        combine(RH / KSPLIT, RB, 3 * RH, bhh1, 0, gh, nullptr);
        gru_pointwise_kernel<<<(RB * RH) / 256, 256, 0, stream>>>(gi, gh, h1, h, h_bf);

        // pv_new = h2 @ Wout + b_out            M=128 N=2048 K=1024, SK=4
        gemm_sk(h_bf, WoutTb, RB, RV, RH);
        combine(RH / KSPLIT, RB, RV, b_out, 0, pv, pv_bf);

        // loss_t = sum((pv_new - y[T-1-t])^2)
        const float* yt = y + (size_t)(RT - 1 - t) * RB * RV;
        loss_partial_kernel<<<(RB * RV) / 256, 256, 0, stream>>>(pv, yt, partials);
        loss_accum_kernel<<<1, 256, 0, stream>>>(partials, accum);
    }

    write_out_kernel<<<1, 1, 0, stream>>>(accum, (float*)d_out);
}
